// SplineConvNet_46712064311585
// MI455X (gfx1250) — compile-verified, matrix operand order checked by reading, not run on hardware
//
#include <hip/hip_runtime.h>
#include <hip/hip_bf16.h>

typedef __attribute__((ext_vector_type(16))) _Float16 v16h;
typedef __attribute__((ext_vector_type(8)))  _Float16 v8h;
typedef __attribute__((ext_vector_type(8)))  float    v8f;

// ---------------------------------------------------------------------------
// f32 -> f16 convert
// ---------------------------------------------------------------------------
__global__ void cvt_f32_f16(const float* __restrict__ in, _Float16* __restrict__ out, long n) {
  long i = (long)blockIdx.x * blockDim.x + threadIdx.x;
  if (i < n) out[i] = (_Float16)in[i];
}

__global__ void zero_f32(float* __restrict__ p, long n) {
  long i = (long)blockIdx.x * blockDim.x + threadIdx.x;
  if (i < n) p[i] = 0.0f;
}

// ---------------------------------------------------------------------------
// Repack W[k][i][o] (f32, row-major) into per-lane B fragments for
// v_wmma_f32_16x16x32_f16.  Block b = (k*ct4 + ct)*ks2 + ks holds 32 lanes x
// 16 halves (1 KB).  Lane's column = ct*16 + (lane&15); half h maps to
// K = ks*32 + (h>>3)*16 + (lane>>4)*8 + (h&7)  (ISA 16-bit 16x32 layout).
// ---------------------------------------------------------------------------
__global__ void repackW(const float* __restrict__ W, _Float16* __restrict__ WB,
                        int IN, int OUT) {
  int blk = blockIdx.x;
  int ks2 = IN >> 5;
  int ct4 = OUT >> 4;
  int ks = blk % ks2;
  int t  = blk / ks2;
  int ct = t % ct4;
  int k  = t / ct4;
  int lane = threadIdx.x >> 4;   // 0..31
  int h    = threadIdx.x & 15;   // 0..15
  int hi   = lane >> 4;
  int col  = ct * 16 + (lane & 15);
  int i    = ks * 32 + ((h >> 3) << 4) + hi * 8 + (h & 7);
  WB[(size_t)blk * 512 + lane * 16 + h] =
      (_Float16)W[((size_t)k * IN + i) * OUT + col];
}

// ---------------------------------------------------------------------------
// Y[n, kk*OUT + o] = sum_i Xh[n,i] * W[k0+kk][i][o]   via WMMA f16->f32.
// grid.x = ceil(N/16) row tiles, grid.y = nk * (OUT/16) col tiles, block = 32.
// ---------------------------------------------------------------------------
__global__ void __launch_bounds__(32)
gemmY(const _Float16* __restrict__ Xh, const _Float16* __restrict__ WB,
      float* __restrict__ Y, int N, int IN, int OUT, int k0, int nk) {
  const int lane = threadIdx.x;
  const int m  = lane & 15;
  const int hi = lane >> 4;
  const int ct4 = OUT >> 4;
  const int kk = (int)blockIdx.y / ct4;    // chunk-local kernel index
  const int ct = (int)blockIdx.y % ct4;
  const int k  = k0 + kk;
  const int ks2 = IN >> 5;
  const int rowTile = blockIdx.x;

  int row  = rowTile * 16 + m;
  int rowc = row < N ? row : N - 1;
  const _Float16* xrow  = Xh + (size_t)rowc * IN + hi * 8;
  const _Float16* wbase = WB + ((size_t)(k * ct4 + ct) * ks2) * 512 + lane * 16;

  v8f acc = {};
  for (int ks = 0; ks < ks2; ++ks) {
    v8h a0 = *(const v8h*)(xrow + ks * 32);
    v8h a1 = *(const v8h*)(xrow + ks * 32 + 16);
    v16h a;
#pragma unroll
    for (int j = 0; j < 8; ++j) { a[j] = a0[j]; a[j + 8] = a1[j]; }
    v16h b = *(const v16h*)(wbase + (size_t)ks * 512);
    acc = __builtin_amdgcn_wmma_f32_16x16x32_f16(false, a, false, b,
                                                 (short)0, acc, false, false);
  }

  const int yStride = nk * OUT;
  const int col = kk * OUT + ct * 16 + m;
#pragma unroll
  for (int r = 0; r < 8; ++r) {
    int orow = rowTile * 16 + r + hi * 8;
    if (orow < N) Y[(size_t)orow * yStride + col] = acc[r];
  }
}

// ---------------------------------------------------------------------------
// Edge pass: 64 threads per edge (one per output channel).  Computes the
// degree-1 B-spline basis inline, gathers up to 4 rows of Y[src], and
// scatter-adds into agg[dst] with f32 global atomics.
// ---------------------------------------------------------------------------
__global__ void edge_scatter(const float* __restrict__ Y,
                             const int* __restrict__ src,
                             const int* __restrict__ dst,
                             const float* __restrict__ attr,
                             float* __restrict__ agg,
                             int E, int k0, int k1, int yStride) {
  long t = (long)blockIdx.x * blockDim.x + threadIdx.x;
  int e = (int)(t >> 6);
  int o = (int)(t & 63);
  if (e >= E) return;

  float vx = attr[2 * e]     * 4.0f;
  float vy = attr[2 * e + 1] * 4.0f;
  float i0x = floorf(vx), i0y = floorf(vy);
  float fx = vx - i0x, fy = vy - i0y;
  int ix = (int)i0x, iy = (int)i0y;

  const float* ybase = Y + (size_t)src[e] * yStride;
  float acc = 0.0f;
  bool any = false;
#pragma unroll
  for (int s = 0; s < 4; ++s) {
    int bx = s & 1, by = s >> 1;
    float w = (bx ? fx : 1.0f - fx) * (by ? fy : 1.0f - fy);
    int kx = min(ix + bx, 4);
    int ky = min(iy + by, 4);
    int kk = kx + 5 * ky;
    if (kk >= k0 && kk < k1) {
      acc += w * ybase[(kk - k0) * 64 + o];
      any = true;
    }
  }
  if (any) atomicAdd(agg + (size_t)dst[e] * 64 + o, acc);
}

// ---------------------------------------------------------------------------
// h[n,o] = relu(agg[n,o] + (x @ root)[n,o] + bias[o]); optionally emit f16
// copy for the next layer's WMMA A operand.  grid = (N/16, OUT/16), block=32.
// ---------------------------------------------------------------------------
__global__ void __launch_bounds__(32)
combineK(const _Float16* __restrict__ Xh, const _Float16* __restrict__ rootB,
         const float* __restrict__ agg, const float* __restrict__ bias,
         float* __restrict__ hOut, _Float16* __restrict__ XhNext,
         int N, int IN, int OUT) {
  const int lane = threadIdx.x;
  const int m  = lane & 15;
  const int hi = lane >> 4;
  const int rowTile = blockIdx.x;
  const int ct = blockIdx.y;
  const int ks2 = IN >> 5;

  int row  = rowTile * 16 + m;
  int rowc = row < N ? row : N - 1;
  const _Float16* xrow  = Xh + (size_t)rowc * IN + hi * 8;
  const _Float16* wbase = rootB + (size_t)(ct * ks2) * 512 + lane * 16;

  v8f acc = {};
  for (int ks = 0; ks < ks2; ++ks) {
    v8h a0 = *(const v8h*)(xrow + ks * 32);
    v8h a1 = *(const v8h*)(xrow + ks * 32 + 16);
    v16h a;
#pragma unroll
    for (int j = 0; j < 8; ++j) { a[j] = a0[j]; a[j + 8] = a1[j]; }
    v16h b = *(const v16h*)(wbase + (size_t)ks * 512);
    acc = __builtin_amdgcn_wmma_f32_16x16x32_f16(false, a, false, b,
                                                 (short)0, acc, false, false);
  }

  int col = ct * 16 + m;
  float bv = bias[col];
#pragma unroll
  for (int r = 0; r < 8; ++r) {
    int orow = rowTile * 16 + r + hi * 8;
    if (orow < N) {
      float v = acc[r] + agg[(size_t)orow * OUT + col] + bv;
      v = v > 0.0f ? v : 0.0f;
      hOut[(size_t)orow * OUT + col] = v;
      if (XhNext) XhNext[(size_t)orow * OUT + col] = (_Float16)v;
    }
  }
}

// ---------------------------------------------------------------------------
// Final FC: out[n,0:3] = h[n,:] @ fc_w + fc_b
// ---------------------------------------------------------------------------
__global__ void fcK(const float* __restrict__ h, const float* __restrict__ w,
                    const float* __restrict__ b, float* __restrict__ out, int N) {
  int n = blockIdx.x * blockDim.x + threadIdx.x;
  if (n >= N) return;
  float a0 = b[0], a1 = b[1], a2 = b[2];
  const float* hr = h + (size_t)n * 64;
#pragma unroll 8
  for (int o = 0; o < 64; ++o) {
    float xv = hr[o];
    a0 += xv * w[o * 3 + 0];
    a1 += xv * w[o * 3 + 1];
    a2 += xv * w[o * 3 + 2];
  }
  out[(size_t)n * 3 + 0] = a0;
  out[(size_t)n * 3 + 1] = a1;
  out[(size_t)n * 3 + 2] = a2;
}

// ---------------------------------------------------------------------------
// Host launcher
// ---------------------------------------------------------------------------
extern "C" void kernel_launch(void* const* d_in, const int* in_sizes, int n_in,
                              void* d_out, int out_size, void* d_ws, size_t ws_size,
                              hipStream_t stream) {
  const float* x  = (const float*)d_in[0];
  const int*   ei = (const int*)d_in[1];
  const float* ea = (const float*)d_in[2];
  const float* W1 = (const float*)d_in[3];
  const float* r1 = (const float*)d_in[4];
  const float* b1 = (const float*)d_in[5];
  const float* W2 = (const float*)d_in[6];
  const float* r2 = (const float*)d_in[7];
  const float* b2 = (const float*)d_in[8];
  const float* fw = (const float*)d_in[9];
  const float* fb = (const float*)d_in[10];

  const int N = in_sizes[0] / 32;   // 50000
  const int E = in_sizes[2] / 2;    // 400000
  const int* src = ei;
  const int* dst = ei + E;

  char* ws = (char*)d_ws;
  size_t off = 0;
  auto alloc = [&](size_t bytes) {
    size_t o = off;
    off = (off + bytes + 255) & ~(size_t)255;
    return o;
  };
  _Float16* XhA   = (_Float16*)(ws + alloc((size_t)N * 32 * 2)); // layer-1 acts (f16)
  _Float16* XhB   = (_Float16*)(ws + alloc((size_t)N * 64 * 2)); // layer-2 acts (f16)
  _Float16* WB    = (_Float16*)(ws + alloc((size_t)200 * 512 * 2)); // repacked W (max 25*4*2 blocks)
  _Float16* rootB = (_Float16*)(ws + alloc((size_t)8 * 512 * 2));   // repacked root
  float*    agg   = (float*)(ws + alloc((size_t)N * 64 * 4));
  float*    h     = (float*)(ws + alloc((size_t)N * 64 * 4));

  // Remaining workspace holds Y chunks; chunk as many kernel slices as fit.
  size_t perK   = (size_t)N * 64 * 4;
  size_t yAvail = ws_size > off ? ws_size - off : 0;
  int kcap = (int)(yAvail / perK);
  if (kcap < 1) kcap = 1;
  if (kcap > 25) kcap = 25;
  float* Y = (float*)(ws + off);

  const int rowTiles = (N + 15) / 16;
  const long aggN = (long)N * 64;
  const long edgeThreads = (long)E * 64;

  // ---------------- Layer 1 (IN=32 -> OUT=64) ----------------
  {
    long nx = (long)N * 32;
    cvt_f32_f16<<<(unsigned)((nx + 255) / 256), 256, 0, stream>>>(x, XhA, nx);
    repackW<<<25 * 4 * 1, 512, 0, stream>>>(W1, WB, 32, 64);
    repackW<<<1 * 4 * 1, 512, 0, stream>>>(r1, rootB, 32, 64);
    zero_f32<<<(unsigned)((aggN + 255) / 256), 256, 0, stream>>>(agg, aggN);
    for (int k0 = 0; k0 < 25; k0 += kcap) {
      int k1 = k0 + kcap < 25 ? k0 + kcap : 25;
      int nk = k1 - k0;
      dim3 g(rowTiles, nk * 4);
      gemmY<<<g, 32, 0, stream>>>(XhA, WB, Y, N, 32, 64, k0, nk);
      edge_scatter<<<(unsigned)((edgeThreads + 255) / 256), 256, 0, stream>>>(
          Y, src, dst, ea, agg, E, k0, k1, nk * 64);
    }
    combineK<<<dim3(rowTiles, 4), 32, 0, stream>>>(XhA, rootB, agg, b1, h, XhB,
                                                   N, 32, 64);
  }

  // ---------------- Layer 2 (IN=64 -> OUT=64) ----------------
  {
    repackW<<<25 * 4 * 2, 512, 0, stream>>>(W2, WB, 64, 64);
    repackW<<<1 * 4 * 2, 512, 0, stream>>>(r2, rootB, 64, 64);
    zero_f32<<<(unsigned)((aggN + 255) / 256), 256, 0, stream>>>(agg, aggN);
    for (int k0 = 0; k0 < 25; k0 += kcap) {
      int k1 = k0 + kcap < 25 ? k0 + kcap : 25;
      int nk = k1 - k0;
      dim3 g(rowTiles, nk * 4);
      gemmY<<<g, 32, 0, stream>>>(XhB, WB, Y, N, 64, 64, k0, nk);
      edge_scatter<<<(unsigned)((edgeThreads + 255) / 256), 256, 0, stream>>>(
          Y, src, dst, ea, agg, E, k0, k1, nk * 64);
    }
    combineK<<<dim3(rowTiles, 4), 32, 0, stream>>>(XhB, rootB, agg, b2, h,
                                                   (_Float16*)nullptr, N, 64, 64);
  }

  // ---------------- Final FC ----------------
  fcK<<<(N + 255) / 256, 256, 0, stream>>>(h, fw, fb, (float*)d_out, N);
}
